// KDABlock_77945066487881
// MI455X (gfx1250) — matrix operand order repaired and checked by reading, not compiled, not run on hardware
//
#include <hip/hip_runtime.h>
#include <hip/hip_bf16.h>
#include <math.h>

// ---------------- problem constants ----------------
#define C_B   4
#define C_T   1024
#define C_D   1024
#define C_H   8
#define C_DK  128
#define C_DV  128
#define C_R   64
#define C_DFF 4096
#define C_K   4
#define C_M   (C_B * C_T)   // 4096 token rows
#define C_DECAY 0.9f
#define C_THR   0.5f
#define C_EPS   1e-6f

typedef __bf16 bf16;
typedef __attribute__((ext_vector_type(16))) __bf16 v16bf;
typedef __attribute__((ext_vector_type(8)))  float  v8f;

__device__ __forceinline__ float siluf(float x) { return x / (1.f + __expf(-x)); }
__device__ __forceinline__ float sigmf(float x) { return 1.f / (1.f + __expf(-x)); }

// ---------------- f32 -> bf16 convert ----------------
__global__ void k_f2bf(const float* __restrict__ in, bf16* __restrict__ out, size_t n) {
    size_t i = (size_t)blockIdx.x * 256 + threadIdx.x;
    if (i < n) out[i] = (bf16)in[i];
}

// ---------------- RMSNorm over last dim (Dn), one block per row ----------------
__global__ void k_rms(const float* __restrict__ x, const float* __restrict__ w,
                      float* __restrict__ outF, bf16* __restrict__ outB, int Dn) {
    int row = blockIdx.x;
    const float* xr = x + (size_t)row * Dn;
    __shared__ float red[256];
    float ss = 0.f;
    for (int i = threadIdx.x; i < Dn; i += 256) { float v = xr[i]; ss += v * v; }
    red[threadIdx.x] = ss;
    __syncthreads();
    for (int s = 128; s > 0; s >>= 1) {
        if (threadIdx.x < s) red[threadIdx.x] += red[threadIdx.x + s];
        __syncthreads();
    }
    float inv = rsqrtf(red[0] / (float)Dn + C_EPS);
    for (int i = threadIdx.x; i < Dn; i += 256) {
        float v = xr[i] * w[i] * inv;
        if (outF) outF[(size_t)row * Dn + i] = v;
        if (outB) outB[(size_t)row * Dn + i] = (bf16)v;
    }
}

// ---------------- causal depthwise conv (K=4) + SiLU ----------------
__global__ void k_conv_silu(const float* __restrict__ t0, const float* __restrict__ cw,
                            const float* __restrict__ cb, float* __restrict__ hf,
                            bf16* __restrict__ hb) {
    size_t idx = (size_t)blockIdx.x * 256 + threadIdx.x;
    size_t total = (size_t)C_M * C_D;
    if (idx >= total) return;
    int d = (int)(idx % C_D);
    size_t bt = idx / C_D;
    int t = (int)(bt % C_T);
    int b = (int)(bt / C_T);
    float acc = cb[d];
#pragma unroll
    for (int j = 0; j < C_K; j++) {
        int ts = t - (C_K - 1) + j;
        if (ts >= 0) acc += t0[((size_t)b * C_T + ts) * C_D + d] * cw[d * C_K + j];
    }
    float s = siluf(acc);
    hf[idx] = s;
    hb[idx] = (bf16)s;
}

// ---------------- WMMA bf16 GEMM: C[M,N] = A[M,K] * W[N,K]^T ----------------
// block = 256 threads = 8 waves arranged 4(M) x 2(N); each wave owns a 32x64
// output tile = 2x4 accumulators sharing 2 A-frags / 4 B-frags per k-step
// (8 v_wmma per 12 b128-loads). Block tile: 128(M) x 128(N).
// epi: 0 none, 1 silu, 2 sigmoid.  bias[n], resid[m*N+n], f32 C / bf16 Cb optional.
__global__ __launch_bounds__(256)
void k_gemm(const bf16* __restrict__ A, const bf16* __restrict__ W,
            const float* __restrict__ bias, const float* __restrict__ resid,
            float* __restrict__ C, bf16* __restrict__ Cb,
            int M, int N, int Kd, int epi) {
    int lane = threadIdx.x & 31;
    int wave = threadIdx.x >> 5;
    int wm = wave & 3, wn = wave >> 2;
    int m0 = blockIdx.y * 128 + wm * 32;
    int n0 = blockIdx.x * 128 + wn * 64;
    if (n0 >= N || m0 >= M) return;           // wave-uniform: EXEC stays all-ones
    int half = lane >> 4;
    int lr   = lane & 15;
    const bf16* Ar0 = A + (size_t)(m0 + lr) * Kd;
    const bf16* Ar1 = A + (size_t)(m0 + 16 + lr) * Kd;
    const bf16* Wr0 = W + (size_t)(n0 + lr) * Kd;
    const bf16* Wr1 = W + (size_t)(n0 + 16 + lr) * Kd;
    const bf16* Wr2 = W + (size_t)(n0 + 32 + lr) * Kd;
    const bf16* Wr3 = W + (size_t)(n0 + 48 + lr) * Kd;
    v8f acc[2][4] = {};
    for (int k0 = 0; k0 < Kd; k0 += 32) {
        if (k0 + 64 < Kd) {                   // pull next K-chunk toward the WGP
            __builtin_prefetch(Ar0 + k0 + 64, 0, 3);
            __builtin_prefetch(Ar1 + k0 + 64, 0, 3);
            __builtin_prefetch(Wr0 + k0 + 64, 0, 3);
            __builtin_prefetch(Wr1 + k0 + 64, 0, 3);
            __builtin_prefetch(Wr2 + k0 + 64, 0, 3);
            __builtin_prefetch(Wr3 + k0 + 64, 0, 3);
        }
        v16bf a0, a1, b0, b1, b2, b3;
#pragma unroll
        for (int j = 0; j < 16; j++) {
            // A 16x32 bf16 lane layout (contiguous, matches B): k = 16*half + j
            int ka = k0 + 16 * half + j;
            a0[j] = Ar0[ka];
            a1[j] = Ar1[ka];
            // B 32x16 bf16 lane layout: k = 16*half + j
            int kb = k0 + 16 * half + j;
            b0[j] = Wr0[kb]; b1[j] = Wr1[kb]; b2[j] = Wr2[kb]; b3[j] = Wr3[kb];
        }
        acc[0][0] = __builtin_amdgcn_wmma_f32_16x16x32_bf16(false, a0, false, b0, (short)0, acc[0][0], false, false);
        acc[0][1] = __builtin_amdgcn_wmma_f32_16x16x32_bf16(false, a0, false, b1, (short)0, acc[0][1], false, false);
        acc[0][2] = __builtin_amdgcn_wmma_f32_16x16x32_bf16(false, a0, false, b2, (short)0, acc[0][2], false, false);
        acc[0][3] = __builtin_amdgcn_wmma_f32_16x16x32_bf16(false, a0, false, b3, (short)0, acc[0][3], false, false);
        acc[1][0] = __builtin_amdgcn_wmma_f32_16x16x32_bf16(false, a1, false, b0, (short)0, acc[1][0], false, false);
        acc[1][1] = __builtin_amdgcn_wmma_f32_16x16x32_bf16(false, a1, false, b1, (short)0, acc[1][1], false, false);
        acc[1][2] = __builtin_amdgcn_wmma_f32_16x16x32_bf16(false, a1, false, b2, (short)0, acc[1][2], false, false);
        acc[1][3] = __builtin_amdgcn_wmma_f32_16x16x32_bf16(false, a1, false, b3, (short)0, acc[1][3], false, false);
    }
#pragma unroll
    for (int ms = 0; ms < 2; ms++) {
#pragma unroll
        for (int ns = 0; ns < 4; ns++) {
#pragma unroll
            for (int r = 0; r < 8; r++) {
                int m = m0 + ms * 16 + r + 8 * half;
                int n = n0 + ns * 16 + lr;
                float v = acc[ms][ns][r];
                if (bias) v += bias[n];
                if (epi == 1) v = siluf(v);
                else if (epi == 2) v = sigmf(v);
                if (resid) v += resid[(size_t)m * N + n];
                if (C)  C[(size_t)m * N + n] = v;
                if (Cb) Cb[(size_t)m * N + n] = (bf16)v;
            }
        }
    }
}

// ---------------- fused SwiGLU up GEMM: out = bf16(silu(A*W1^T) * (A*W3^T)) ----------------
// 8 waves, each owns a 16x64 tile: 1 A-frag reused by 4 W1-frags + 4 W3-frags
// (8 v_wmma per k-step). Block tile: 128(M) x 64(N).
__global__ __launch_bounds__(256)
void k_gemm_ff13(const bf16* __restrict__ A, const bf16* __restrict__ W1,
                 const bf16* __restrict__ W3, bf16* __restrict__ OutB,
                 int M, int N, int Kd) {
    int lane = threadIdx.x & 31;
    int wave = threadIdx.x >> 5;
    int m0 = blockIdx.y * 128 + wave * 16;
    int n0 = blockIdx.x * 64;
    if (n0 >= N || m0 >= M) return;
    int half = lane >> 4;
    int lr   = lane & 15;
    const bf16* Arow = A + (size_t)(m0 + lr) * Kd;
    const bf16* P1[4], *P3[4];
#pragma unroll
    for (int ns = 0; ns < 4; ns++) {
        P1[ns] = W1 + (size_t)(n0 + ns * 16 + lr) * Kd;
        P3[ns] = W3 + (size_t)(n0 + ns * 16 + lr) * Kd;
    }
    v8f acc1[4] = {}, acc3[4] = {};
    for (int k0 = 0; k0 < Kd; k0 += 32) {
        if (k0 + 64 < Kd) __builtin_prefetch(Arow + k0 + 64, 0, 3);
        v16bf a, b1[4], b3[4];
#pragma unroll
        for (int j = 0; j < 16; j++) {
            int ka = k0 + 16 * half + j;
            a[j] = Arow[ka];
            int kb = k0 + 16 * half + j;
#pragma unroll
            for (int ns = 0; ns < 4; ns++) { b1[ns][j] = P1[ns][kb]; b3[ns][j] = P3[ns][kb]; }
        }
#pragma unroll
        for (int ns = 0; ns < 4; ns++) {
            acc1[ns] = __builtin_amdgcn_wmma_f32_16x16x32_bf16(false, a, false, b1[ns], (short)0, acc1[ns], false, false);
            acc3[ns] = __builtin_amdgcn_wmma_f32_16x16x32_bf16(false, a, false, b3[ns], (short)0, acc3[ns], false, false);
        }
    }
#pragma unroll
    for (int ns = 0; ns < 4; ns++) {
#pragma unroll
        for (int r = 0; r < 8; r++) {
            int m = m0 + r + 8 * half;
            int n = n0 + ns * 16 + lr;
            OutB[(size_t)m * N + n] = (bf16)(siluf(acc1[ns][r]) * acc3[ns][r]);
        }
    }
}

// ---------------- beta GEMV: out[m,h] = dot(h[m,:], beta_w[h,:]) + beta_b[h] ----------------
__global__ void k_beta(const float* __restrict__ hf, const float* __restrict__ bw,
                       const float* __restrict__ bb, float* __restrict__ out) {
    int m = blockIdx.x;
    int g = threadIdx.x >> 5;   // head 0..7
    int l = threadIdx.x & 31;
    const float* hr = hf + (size_t)m * C_D;
    const float* wr = bw + (size_t)g * C_D;
    float p = 0.f;
    for (int kk = l; kk < C_D; kk += 32) p += hr[kk] * wr[kk];
#pragma unroll
    for (int o = 16; o > 0; o >>= 1) p += __shfl_xor(p, o, 32);
    if (l == 0) out[(size_t)m * C_H + g] = p + bb[g];
}

// ---------------- q/k L2-normalize, one block per (b,t,h), 128 threads ----------------
__global__ void k_l2norm_qk(float* __restrict__ q, float* __restrict__ kx) {
    int row = blockIdx.x;
    int tid = threadIdx.x;
    __shared__ float red[128];
    size_t base = (size_t)row * C_DK;
    float qv = q[base + tid];
    red[tid] = qv * qv;
    __syncthreads();
    for (int s = 64; s > 0; s >>= 1) { if (tid < s) red[tid] += red[tid + s]; __syncthreads(); }
    float inv = 1.f / (sqrtf(red[0]) + 1e-6f);
    q[base + tid] = qv * inv;
    __syncthreads();
    float kv = kx[base + tid];
    red[tid] = kv * kv;
    __syncthreads();
    for (int s = 64; s > 0; s >>= 1) { if (tid < s) red[tid] += red[tid + s]; __syncthreads(); }
    inv = 1.f / (sqrtf(red[0]) + 1e-6f);
    kx[base + tid] = kv * inv;
}

// ---------------- LIF + gated delta-rule scan, one block per (b,h) ----------------
__global__ __launch_bounds__(256)
void k_scan(const float* __restrict__ q, const float* __restrict__ kx,
            const float* __restrict__ v, const float* __restrict__ dr,
            const float* __restrict__ abase, const float* __restrict__ bbase,
            const float* __restrict__ st0, const float* __restrict__ mem0,
            const float* __restrict__ a_sp, const float* __restrict__ b_sp,
            float* __restrict__ y) {
    extern __shared__ float sm[];
    float* S     = sm;                 // 128 * 129 (padded pitch: conflict-free)
    float* sQ    = S + 128 * 129;
    float* sK    = sQ + 128;
    float* sV    = sK + 128;
    float* sA    = sV + 128;
    float* sMem  = sA + 128;
    float* sRedA = sMem + 128;         // 256
    float* sRedB = sRedA + 256;        // 256
    float* sSc   = sRedB + 256;        // 2 (+pad)

    int bh = blockIdx.x;
    int b  = bh / C_H;
    int h  = bh % C_H;
    int tid  = threadIdx.x;
    int vcol = tid & 127;
    int half = tid >> 7;

    const float* S0 = st0 + (size_t)bh * C_DK * C_DV;
    for (int kk = 0; kk < 64; kk++) {
        int kr = half * 64 + kk;
        S[kr * 129 + vcol] = S0[(size_t)kr * C_DV + vcol];
    }
    if (tid < 128) sMem[tid] = mem0[(size_t)bh * C_DK + tid];
    __syncthreads();

    for (int t = 0; t < C_T; t++) {
        size_t base = ((size_t)(b * C_T + t) * C_H + h) * C_DK;
        if (tid < 128) {
            int d = tid;
            float m = sMem[d] * C_DECAY + dr[base + d];
            float sp = (m > C_THR) ? 1.f : 0.f;    // forward value of surrogate spike
            sMem[d] = m - sp * C_THR;
            sA[d] = sigmf(abase[base + d] + a_sp[h * C_DK + d] * sp);
            sQ[d] = q[base + d];
            sK[d] = kx[base + d];
            sV[d] = v[base + d];
            sRedA[d] = sp;                          // |spike| sum
            sRedB[d] = sp * b_sp[h * C_DK + d];     // spike . beta_spike
        }
        __syncthreads();
        if (tid < 32) {
            float s1 = sRedA[tid] + sRedA[tid + 32] + sRedA[tid + 64] + sRedA[tid + 96];
            float s2 = sRedB[tid] + sRedB[tid + 32] + sRedB[tid + 64] + sRedB[tid + 96];
#pragma unroll
            for (int o = 16; o > 0; o >>= 1) {
                s1 += __shfl_xor(s1, o, 32);
                s2 += __shfl_xor(s2, o, 32);
            }
            if (tid == 0) { sSc[0] = s1; sSc[1] = s2; }
        }
        __syncthreads();
        float beta = sigmf(bbase[(size_t)(b * C_T + t) * C_H + h] + sSc[1]);
        float coef = (sSc[0] > 0.f) ? beta : 0.f;

        float p = 0.f;
#pragma unroll 4
        for (int kk = 0; kk < 64; kk++) {
            int kr = half * 64 + kk;
            float s = S[kr * 129 + vcol] * sA[kr];
            S[kr * 129 + vcol] = s;
            p += sK[kr] * s;
        }
        sRedA[half * 128 + vcol] = p;
        __syncthreads();

        float delta = coef * (sV[vcol] - (sRedA[vcol] + sRedA[128 + vcol]));
        float o = 0.f;
#pragma unroll 4
        for (int kk = 0; kk < 64; kk++) {
            int kr = half * 64 + kk;
            float s = S[kr * 129 + vcol] + sK[kr] * delta;
            S[kr * 129 + vcol] = s;
            o += sQ[kr] * s;
        }
        sRedB[half * 128 + vcol] = o;
        __syncthreads();
        if (tid < 128) y[base + tid] = sRedB[tid] + sRedB[128 + tid];
        __syncthreads();
    }
}

// ---------------- headwise RMSNorm -> bf16 ----------------
__global__ void k_headnorm(const float* __restrict__ y, const float* __restrict__ w,
                           bf16* __restrict__ out) {
    int row = blockIdx.x;          // (b*T + t)*H + h
    int h   = row % C_H;
    int tid = threadIdx.x;
    __shared__ float red[128];
    size_t base = (size_t)row * C_DV;
    float v = y[base + tid];
    red[tid] = v * v;
    __syncthreads();
    for (int s = 64; s > 0; s >>= 1) { if (tid < s) red[tid] += red[tid + s]; __syncthreads(); }
    float inv = rsqrtf(red[0] / (float)C_DV + C_EPS);
    out[base + tid] = (bf16)(v * w[h * C_DV + tid] * inv);
}

// ---------------- y2 = x + o * gate ----------------
__global__ void k_gate(const float* __restrict__ x, const float* __restrict__ o,
                       const float* __restrict__ g, float* __restrict__ y2, size_t n) {
    size_t i = (size_t)blockIdx.x * 256 + threadIdx.x;
    if (i < n) y2[i] = x[i] + o[i] * g[i];
}

// ================= host orchestration =================
extern "C" void kernel_launch(void* const* d_in, const int* in_sizes, int n_in,
                              void* d_out, int out_size, void* d_ws, size_t ws_size,
                              hipStream_t stream) {
    (void)in_sizes; (void)n_in; (void)out_size; (void)ws_size;
    const float* x       = (const float*)d_in[0];
    const float* state0  = (const float*)d_in[1];
    const float* mem0    = (const float*)d_in[2];
    const float* norm_w  = (const float*)d_in[3];
    const float* conv_w  = (const float*)d_in[4];
    const float* conv_b  = (const float*)d_in[5];
    const float* q_w     = (const float*)d_in[6];
    const float* k_w     = (const float*)d_in[7];
    const float* v_w     = (const float*)d_in[8];
    const float* out_w   = (const float*)d_in[9];
    const float* spike_w = (const float*)d_in[10];
    const float* aup_w   = (const float*)d_in[11];
    const float* aup_b   = (const float*)d_in[12];
    const float* adn_w   = (const float*)d_in[13];
    const float* adn_b   = (const float*)d_in[14];
    const float* a_sp    = (const float*)d_in[15];
    const float* beta_w  = (const float*)d_in[16];
    const float* beta_b  = (const float*)d_in[17];
    const float* b_sp    = (const float*)d_in[18];
    const float* hn_w    = (const float*)d_in[19];
    const float* u1_w    = (const float*)d_in[20];
    const float* u1_b    = (const float*)d_in[21];
    const float* u2_w    = (const float*)d_in[22];
    const float* u2_b    = (const float*)d_in[23];
    const float* ffn_w   = (const float*)d_in[24];
    const float* ff_w1   = (const float*)d_in[25];
    const float* ff_w3   = (const float*)d_in[26];
    const float* ff_w2   = (const float*)d_in[27];

    size_t off = 0;
    auto alloc = [&](size_t bytes) -> char* {
        char* p = (char*)d_ws + off;
        off += (bytes + 255) & ~(size_t)255;
        return p;
    };
    const size_t MD = (size_t)C_M * C_D;

    bf16* Xb   = (bf16*)alloc(MD * 2);
    bf16* Wqb  = (bf16*)alloc((size_t)C_D * C_D * 2);
    bf16* Wkb  = (bf16*)alloc((size_t)C_D * C_D * 2);
    bf16* Wvb  = (bf16*)alloc((size_t)C_D * C_D * 2);
    bf16* Wspb = (bf16*)alloc((size_t)C_D * C_D * 2);
    bf16* Woub = (bf16*)alloc((size_t)C_D * C_D * 2);
    bf16* Waub = (bf16*)alloc((size_t)C_R * C_D * 2);
    bf16* Wadb = (bf16*)alloc((size_t)C_D * C_R * 2);
    bf16* Wu1b = (bf16*)alloc((size_t)C_R * C_D * 2);
    bf16* Wu2b = (bf16*)alloc((size_t)C_D * C_R * 2);
    bf16* Wf1b = (bf16*)alloc((size_t)C_DFF * C_D * 2);
    bf16* Wf3b = (bf16*)alloc((size_t)C_DFF * C_D * 2);
    bf16* Wf2b = (bf16*)alloc((size_t)C_D * C_DFF * 2);

    float* T0   = (float*)alloc(MD * 4);
    float* Hf   = (float*)alloc(MD * 4);
    bf16*  Hb   = (bf16*)alloc(MD * 2);
    float* Q    = (float*)alloc(MD * 4);
    float* Kx   = (float*)alloc(MD * 4);
    float* V    = (float*)alloc(MD * 4);
    float* DR   = (float*)alloc(MD * 4);
    bf16*  ARb  = (bf16*)alloc((size_t)C_M * C_R * 2);
    float* AB   = (float*)alloc(MD * 4);
    float* BBm  = (float*)alloc((size_t)C_M * C_H * 4);
    float* Y    = (float*)alloc(MD * 4);
    bf16*  Yb   = (bf16*)alloc(MD * 2);
    float* O    = (float*)alloc(MD * 4);
    bf16*  G1b  = (bf16*)alloc((size_t)C_M * C_R * 2);
    float* GATE = (float*)alloc(MD * 4);
    float* Y2   = (float*)alloc(MD * 4);
    bf16*  Zb   = (bf16*)alloc(MD * 2);
    bf16*  FH   = (bf16*)alloc((size_t)C_M * C_DFF * 2);

    auto cvt = [&](const float* src, bf16* dst, size_t n) {
        k_f2bf<<<dim3((unsigned)((n + 255) / 256)), dim3(256), 0, stream>>>(src, dst, n);
    };
    auto gemm = [&](const bf16* A, const bf16* W, const float* bias, const float* resid,
                    float* Cc, bf16* Cb, int Mm, int Nn, int Kd, int epi) {
        dim3 g((unsigned)((Nn + 127) / 128), (unsigned)((Mm + 127) / 128));
        k_gemm<<<g, 256, 0, stream>>>(A, W, bias, resid, Cc, Cb, Mm, Nn, Kd, epi);
    };

    // weight + activation converts
    cvt(x, Xb, MD);
    cvt(q_w, Wqb, (size_t)C_D * C_D);
    cvt(k_w, Wkb, (size_t)C_D * C_D);
    cvt(v_w, Wvb, (size_t)C_D * C_D);
    cvt(spike_w, Wspb, (size_t)C_D * C_D);
    cvt(out_w, Woub, (size_t)C_D * C_D);
    cvt(aup_w, Waub, (size_t)C_R * C_D);
    cvt(adn_w, Wadb, (size_t)C_D * C_R);
    cvt(u1_w, Wu1b, (size_t)C_R * C_D);
    cvt(u2_w, Wu2b, (size_t)C_D * C_R);
    cvt(ff_w1, Wf1b, (size_t)C_DFF * C_D);
    cvt(ff_w3, Wf3b, (size_t)C_DFF * C_D);
    cvt(ff_w2, Wf2b, (size_t)C_D * C_DFF);

    // input RMSNorm -> T0 (f32)
    k_rms<<<dim3(C_M), dim3(256), 0, stream>>>(x, norm_w, T0, (bf16*)nullptr, C_D);
    // conv + silu -> Hf (f32), Hb (bf16)
    k_conv_silu<<<dim3((unsigned)((MD + 255) / 256)), dim3(256), 0, stream>>>(T0, conv_w, conv_b, Hf, Hb);

    // projections (WMMA)
    gemm(Hb, Wqb,  nullptr, nullptr, Q,  nullptr, C_M, C_D, C_D, 0);
    gemm(Hb, Wkb,  nullptr, nullptr, Kx, nullptr, C_M, C_D, C_D, 0);
    gemm(Hb, Wvb,  nullptr, nullptr, V,  nullptr, C_M, C_D, C_D, 0);
    gemm(Hb, Wspb, nullptr, nullptr, DR, nullptr, C_M, C_D, C_D, 0);
    // alpha low-rank: silu(h@aup^T + b) (bf16 direct) then @ adn^T + b
    gemm(Hb, Waub, aup_b, nullptr, nullptr, ARb, C_M, C_R, C_D, 1);
    gemm(ARb, Wadb, adn_b, nullptr, AB, nullptr, C_M, C_D, C_R, 0);
    // beta base (N=8 GEMV)
    k_beta<<<dim3(C_M), dim3(256), 0, stream>>>(Hf, beta_w, beta_b, BBm);
    // q/k L2 normalize
    k_l2norm_qk<<<dim3(C_M * C_H), dim3(128), 0, stream>>>(Q, Kx);

    // sequential LIF + delta-rule scan
    size_t smem = (size_t)(128 * 129 + 5 * 128 + 256 + 256 + 8) * sizeof(float);
    k_scan<<<dim3(C_B * C_H), dim3(256), smem, stream>>>(Q, Kx, V, DR, AB, BBm,
                                                         state0, mem0, a_sp, b_sp, Y);

    // head RMSNorm -> bf16, out projection
    k_headnorm<<<dim3(C_M * C_H), dim3(128), 0, stream>>>(Y, hn_w, Yb);
    gemm(Yb, Woub, nullptr, nullptr, O, nullptr, C_M, C_D, C_D, 0);

    // gate: sigmoid(silu(x@u1^T+b)@u2^T+b)
    gemm(Xb, Wu1b, u1_b, nullptr, nullptr, G1b, C_M, C_R, C_D, 1);
    gemm(G1b, Wu2b, u2_b, nullptr, GATE, nullptr, C_M, C_D, C_R, 2);
    k_gate<<<dim3((unsigned)((MD + 255) / 256)), dim3(256), 0, stream>>>(x, O, GATE, Y2, MD);

    // FFN: z = rms(y2); FH = silu(z@w1^T) * (z@w3^T); out = y2 + FH @ w2^T
    k_rms<<<dim3(C_M), dim3(256), 0, stream>>>(Y2, ffn_w, (float*)nullptr, Zb, C_D);
    {
        dim3 g((unsigned)(C_DFF / 64), (unsigned)(C_M / 128));
        k_gemm_ff13<<<g, 256, 0, stream>>>(Zb, Wf1b, Wf3b, FH, C_M, C_DFF, C_D);
    }
    gemm(FH, Wf2b, nullptr, Y2, (float*)d_out, nullptr, C_M, C_D, C_DFF, 0);
}
